// DeepseekMoE_82437602279914
// MI455X (gfx1250) — matrix operand order repaired and checked by reading, not compiled, not run on hardware
//
#include <hip/hip_runtime.h>
#include <hip/hip_bf16.h>

// ---------------------------------------------------------------------------
// DeepSeek-style MoE for MI455X (gfx1250), bf16 WMMA pipeline.
//   T tokens, H hidden, I intermediate, E=8 experts, top-2 routing.
//   Stages: zero counts -> x to bf16 -> gate (softmax/top2 -> per-expert
//   lists) -> shared FFN (SwiGLU, writes d_out) -> per expert: FFN1 (SwiGLU
//   into act) + FFN2 (scatter-accumulate into d_out with combine weight).
// ---------------------------------------------------------------------------

typedef __attribute__((ext_vector_type(16))) __bf16 v16bf;
typedef __attribute__((ext_vector_type(8)))  __bf16 v8bf;
typedef __attribute__((ext_vector_type(8)))  float  v8f;

#define N_EXPERTS 8

static __device__ inline unsigned int pack_bf16(float a, float b) {
    __bf16 x = (__bf16)a;
    __bf16 y = (__bf16)b;
    unsigned short ux = __builtin_bit_cast(unsigned short, x);
    unsigned short uy = __builtin_bit_cast(unsigned short, y);
    return (unsigned int)ux | ((unsigned int)uy << 16);
}

// ---------------------------------------------------------------------------
// init: zero per-expert token counters
// ---------------------------------------------------------------------------
__global__ __launch_bounds__(32) void zero_counts_kernel(int* counts) {
    if (threadIdx.x < N_EXPERTS) counts[threadIdx.x] = 0;
}

// ---------------------------------------------------------------------------
// f32 -> bf16 conversion (4 elements / thread)
// ---------------------------------------------------------------------------
__global__ __launch_bounds__(256) void cvt_bf16_kernel(const float* __restrict__ x,
                                                       __bf16* __restrict__ y,
                                                       long long n) {
    long long i = ((long long)blockIdx.x * blockDim.x + threadIdx.x) * 4;
    if (i + 3 < n) {
        float4 f = *(const float4*)(x + i);
        uint2 o;
        o.x = pack_bf16(f.x, f.y);
        o.y = pack_bf16(f.z, f.w);
        *(uint2*)(y + i) = o;
    }
}

// ---------------------------------------------------------------------------
// Router: one wave32 per token. Lane-parallel dot products with the 8 router
// rows, cross-lane reduce, softmax, top-2, renormalize, append to the
// per-expert gather lists via atomics.
// ---------------------------------------------------------------------------
__global__ __launch_bounds__(256) void gate_kernel(const float* __restrict__ x,
                                                   const float* __restrict__ gw,
                                                   int H, int T,
                                                   int* __restrict__ counts,
                                                   int* __restrict__ idxList,
                                                   float* __restrict__ wgtList) {
    int wave = blockIdx.x * (blockDim.x >> 5) + (threadIdx.x >> 5);
    int lane = threadIdx.x & 31;
    if (wave >= T) return;

    const float* xr = x + (size_t)wave * H;
    float acc[N_EXPERTS];
#pragma unroll
    for (int e = 0; e < N_EXPERTS; ++e) acc[e] = 0.0f;

    for (int h = lane; h < H; h += 32) {
        float xv = xr[h];
#pragma unroll
        for (int e = 0; e < N_EXPERTS; ++e) acc[e] += xv * gw[(size_t)e * H + h];
    }
#pragma unroll
    for (int e = 0; e < N_EXPERTS; ++e) {
#pragma unroll
        for (int off = 16; off > 0; off >>= 1)
            acc[e] += __shfl_xor(acc[e], off, 32);
    }

    // softmax over 8 logits (all lanes hold identical values now)
    float mx = acc[0];
#pragma unroll
    for (int e = 1; e < N_EXPERTS; ++e) mx = fmaxf(mx, acc[e]);
    float p[N_EXPERTS];
    float se = 0.0f;
#pragma unroll
    for (int e = 0; e < N_EXPERTS; ++e) { p[e] = __expf(acc[e] - mx); se += p[e]; }
    float inv = 1.0f / se;
#pragma unroll
    for (int e = 0; e < N_EXPERTS; ++e) p[e] *= inv;

    // top-2
    int i0 = 0;
#pragma unroll
    for (int e = 1; e < N_EXPERTS; ++e) if (p[e] > p[i0]) i0 = e;
    int i1 = (i0 == 0) ? 1 : 0;
#pragma unroll
    for (int e = 0; e < N_EXPERTS; ++e)
        if (e != i0 && p[e] > p[i1]) i1 = e;

    float w0 = p[i0], w1 = p[i1];
    float rs = 1.0f / (w0 + w1 + 1e-20f);
    w0 *= rs; w1 *= rs;

    if (lane == 0) {
        int pos0 = atomicAdd(&counts[i0], 1);
        idxList[(size_t)i0 * T + pos0] = wave;
        wgtList[(size_t)i0 * T + pos0] = w0;
        int pos1 = atomicAdd(&counts[i1], 1);
        idxList[(size_t)i1 * T + pos1] = wave;
        wgtList[(size_t)i1 * T + pos1] = w1;
    }
}

// ---------------------------------------------------------------------------
// Stage a 32(K) x 64(N) weight tile into LDS as pair-packed bf16:
//   lds[n*16 + v] = {B(k0+2v, n0+n), B(k0+2v+1, n0+n)}  packed in one dword.
// BNMAJOR=0: element(k,n) = B[k*ldb + n]   (w1 / w2 layout, K-major rows)
// BNMAJOR=1: element(k,n) = B[n*ldb + k]   (sw1 / sw2 / sw3 layout)
// ---------------------------------------------------------------------------
template <int BNMAJOR>
static __device__ inline void stage_b_tile(unsigned int* __restrict__ lds,
                                           const float* __restrict__ Bp,
                                           int ldb, int n0, int k0, int tid,
                                           int Ktot) {
    int n  = tid & 63;
    int vb = (tid >> 6) << 2;  // 0,4,8,12
    unsigned int u0, u1, u2, u3;
    if (BNMAJOR) {
        const float* s = Bp + (size_t)(n0 + n) * ldb + (k0 + 2 * vb);
        float4 f0 = *(const float4*)s;
        float4 f1 = *(const float4*)(s + 4);
        u0 = pack_bf16(f0.x, f0.y); u1 = pack_bf16(f0.z, f0.w);
        u2 = pack_bf16(f1.x, f1.y); u3 = pack_bf16(f1.z, f1.w);
        if (k0 + 32 < Ktot) __builtin_prefetch(s + 32, 0, 1);
    } else {
        int k = k0 + 2 * vb;
        const float* r0 = Bp + (size_t)k * ldb + (n0 + n);
        u0 = pack_bf16(r0[0],                 r0[(size_t)ldb]);
        u1 = pack_bf16(r0[2 * (size_t)ldb],   r0[3 * (size_t)ldb]);
        u2 = pack_bf16(r0[4 * (size_t)ldb],   r0[5 * (size_t)ldb]);
        u3 = pack_bf16(r0[6 * (size_t)ldb],   r0[7 * (size_t)ldb]);
        if (k0 + 32 < Ktot)
            __builtin_prefetch(Bp + (size_t)(k0 + 32 + 2 * vb) * ldb + (n0 + n), 0, 1);
    }
    *(uint4*)&lds[n * 16 + vb] = make_uint4(u0, u1, u2, u3);
}

// A fragment, 16-bit A 16x32 in this file's contiguous k order (k = 16*half + s), not ISA table order:
//   lanes 0-15 (M=lane):    VGPR0-3 = K k0..k0+7,   VGPR4-7 = K k0+8..k0+15
//   lanes 16-31 (M=lane-16): same +16  (koff = 16)
static __device__ inline v16bf load_a_frag(const __bf16* __restrict__ Arow,
                                           int k0, int koff) {
    v8bf lo = *(const v8bf*)(Arow + k0 + koff);
    v8bf hi = *(const v8bf*)(Arow + k0 + koff + 8);
    v16bf a;
#pragma unroll
    for (int i = 0; i < 8; ++i) { a[i] = lo[i]; a[8 + i] = hi[i]; }
    return a;
}

// ---------------------------------------------------------------------------
// WMMA GEMM, 128(M) x 64(N) block tile, 8 waves (each wave: 16 rows x 64 cols),
// K-step 32, bf16 inputs, f32 accumulation.
//   EPI 0: fused SwiGLU  act[r] = silu(A@B) * (A@B2)          (bf16 out)
//   EPI 1: scatter-add   out[outMap[r]] += wRow[r] * (A@B)    (f32 rmw)
//   EPI 2: store         out[r] = A@B                          (f32)
// rowMap: optional gather of A rows (token lists). cntPtr: device row count.
// ---------------------------------------------------------------------------
template <int BNMAJOR, int EPI>
__global__ __launch_bounds__(256) void moe_gemm_kernel(
    const __bf16* __restrict__ A, int lda, int Ktot,
    const int* __restrict__ rowMap,
    const float* __restrict__ B, int ldb,
    const float* __restrict__ B2,
    __bf16* __restrict__ actOut, int ldact,
    float* __restrict__ Cout, int ldc,
    const int* __restrict__ outMap,
    const float* __restrict__ wRow,
    const int* __restrict__ cntPtr, int Mtot) {

    __shared__ unsigned int ldsB[2][64 * 16];  // 8 KB

    int cnt = cntPtr ? *cntPtr : Mtot;
    int m0 = blockIdx.x * 128;
    if (m0 >= cnt) return;
    int n0 = blockIdx.y * 64;

    int tid  = threadIdx.x;
    int lane = tid & 31;
    int wave = tid >> 5;
    int hi   = lane >> 4;    // lane half
    int lml  = lane & 15;
    int koff = hi * 16;

    // A row for this lane (fragment layout: lanes 0-15 / 16-31 both cover M 0-15)
    int mrow = m0 + wave * 16 + lml;
    int rA = mrow < cnt ? mrow : cnt - 1;
    const __bf16* Arow;
    if (rowMap) {
        int t = rowMap[rA];
        if (t < 0) t = 0;
        if (t >= Mtot) t = Mtot - 1;
        Arow = A + (size_t)t * lda;
    } else {
        Arow = A + (size_t)rA * lda;
    }

    v8f acc[4]  = {};
    v8f accU[4] = {};

    for (int k0 = 0; k0 < Ktot; k0 += 32) {
        __syncthreads();
        stage_b_tile<BNMAJOR>(&ldsB[0][0], B, ldb, n0, k0, tid, Ktot);
        if (EPI == 0)
            stage_b_tile<BNMAJOR>(&ldsB[1][0], B2, ldb, n0, k0, tid, Ktot);
        __syncthreads();

        v16bf a = load_a_frag(Arow, k0, koff);
#pragma unroll
        for (int nf = 0; nf < 4; ++nf) {
            // B fragment: lanes 0-15 pairs 0..7 (K 0..15), lanes 16-31 pairs 8..15
            v16bf b = *(const v16bf*)&ldsB[0][(nf * 16 + lml) * 16 + hi * 8];
            acc[nf] = __builtin_amdgcn_wmma_f32_16x16x32_bf16(
                false, a, false, b, (short)0, acc[nf], false, false);
        }
        if (EPI == 0) {
#pragma unroll
            for (int nf = 0; nf < 4; ++nf) {
                v16bf b = *(const v16bf*)&ldsB[1][(nf * 16 + lml) * 16 + hi * 8];
                accU[nf] = __builtin_amdgcn_wmma_f32_16x16x32_bf16(
                    false, a, false, b, (short)0, accU[nf], false, false);
            }
        }
    }

    // C/D layout: lane owns column (lane&15); VGPR v -> row v + 8*(lane>=16)
    int colBase = n0 + lml;
#pragma unroll
    for (int nf = 0; nf < 4; ++nf) {
        int col = colBase + nf * 16;
#pragma unroll
        for (int v = 0; v < 8; ++v) {
            int r = m0 + wave * 16 + v + hi * 8;
            if (r >= cnt) continue;
            if (EPI == 0) {
                float g = acc[nf][v];
                float uu = accU[nf][v];
                float s = g / (1.0f + __expf(-g));
                actOut[(size_t)r * ldact + col] = (__bf16)(s * uu);
            } else if (EPI == 1) {
                int t = outMap[r];
                float w = wRow[r];
                float* p = Cout + (size_t)t * ldc + col;
                *p += w * acc[nf][v];
            } else {
                Cout[(size_t)r * ldc + col] = acc[nf][v];
            }
        }
    }
}

// ---------------------------------------------------------------------------
// host launcher
// ---------------------------------------------------------------------------
static inline size_t align256(size_t v) { return (v + 255) & ~(size_t)255; }

extern "C" void kernel_launch(void* const* d_in, const int* in_sizes, int n_in,
                              void* d_out, int out_size, void* d_ws, size_t ws_size,
                              hipStream_t stream) {
    (void)n_in; (void)out_size; (void)ws_size;

    const float* x   = (const float*)d_in[0];  // [T, H]
    const float* gw  = (const float*)d_in[1];  // [E, H]
    const float* w1  = (const float*)d_in[2];  // [E, H, 2I]
    const float* w2  = (const float*)d_in[3];  // [E, I, H]
    const float* sw1 = (const float*)d_in[4];  // [I, H]
    const float* sw2 = (const float*)d_in[5];  // [I, H]
    const float* sw3 = (const float*)d_in[6];  // [H, I]
    float* out = (float*)d_out;                // [T, H]

    const int E = N_EXPERTS;
    const int H = in_sizes[1] / E;
    const int T = in_sizes[0] / H;
    const int I = in_sizes[2] / (E * H * 2);

    // workspace carve-out
    char* p = (char*)d_ws;
    __bf16* xb  = (__bf16*)p; p += align256((size_t)T * H * sizeof(__bf16));
    __bf16* act = (__bf16*)p; p += align256((size_t)T * I * sizeof(__bf16));
    int*   counts  = (int*)p;   p += 256;
    int*   idxList = (int*)p;   p += align256((size_t)E * T * sizeof(int));
    float* wgtList = (float*)p; p += align256((size_t)E * T * sizeof(float));

    const int Mb  = (T + 127) / 128;
    const int NbI = I / 64;
    const int NbH = H / 64;

    zero_counts_kernel<<<1, 32, 0, stream>>>(counts);

    long long nx = (long long)T * H;
    cvt_bf16_kernel<<<(unsigned)((nx / 4 + 255) / 256), 256, 0, stream>>>(x, xb, nx);

    gate_kernel<<<(unsigned)((T + 7) / 8), 256, 0, stream>>>(
        x, gw, H, T, counts, idxList, wgtList);

    // shared expert: act = silu(x @ sw1^T) * (x @ sw2^T); out = act @ sw3^T
    moe_gemm_kernel<1, 0><<<dim3(Mb, NbI), 256, 0, stream>>>(
        xb, H, H, nullptr, sw1, H, sw2, act, I,
        nullptr, 0, nullptr, nullptr, nullptr, T);
    moe_gemm_kernel<1, 2><<<dim3(Mb, NbH), 256, 0, stream>>>(
        act, I, I, nullptr, sw3, I, nullptr, nullptr, 0,
        out, H, nullptr, nullptr, nullptr, T);

    // routed experts (sequential; act buffer reused, out accumulated)
    for (int e = 0; e < E; ++e) {
        const float* w1e = w1 + (size_t)e * H * (2 * I);
        const float* w2e = w2 + (size_t)e * I * H;
        moe_gemm_kernel<0, 0><<<dim3(Mb, NbI), 256, 0, stream>>>(
            xb, H, H, idxList + (size_t)e * T, w1e, 2 * I, w1e + I, act, I,
            nullptr, 0, nullptr, nullptr, counts + e, T);
        moe_gemm_kernel<0, 1><<<dim3(Mb, NbH), 256, 0, stream>>>(
            act, I, I, nullptr, w2e, H, nullptr, nullptr, 0,
            out, H, idxList + (size_t)e * T, wgtList + (size_t)e * T,
            counts + e, T);
    }
}
